// A3TGCNNet_27573690040588
// MI455X (gfx1250) — matrix operand order repaired and checked by reading, not compiled, not run on hardware
//
#include <hip/hip_runtime.h>
#include <hip/hip_bf16.h>

typedef __attribute__((ext_vector_type(16))) _Float16 v16h;
typedef __attribute__((ext_vector_type(8)))  float    v8f;

#define P_DIM 12
#define H_DIM 128
#define O_DIM 12

// ---------------- init: agg = 0, deg = 1 (self-loop weight) ----------------
__global__ void a3t_init_kernel(float* __restrict__ agg, float* __restrict__ deg, int n) {
    int t = blockIdx.x * blockDim.x + threadIdx.x;
    if (t < n * P_DIM) agg[t] = 0.0f;
    if (t < n)         deg[t] = 1.0f;
}

// ---------------- deg[dst] += w (segment_sum of edge weights) ----------------
__global__ void a3t_degree_kernel(const int* __restrict__ eidx, const float* __restrict__ w,
                                  float* __restrict__ deg, int E) {
    int e = blockIdx.x * blockDim.x + threadIdx.x;
    if (e >= E) return;
    atomicAdd(&deg[eidx[E + e]], w[e]);
}

// ---------------- dinv = deg > 0 ? rsqrt(deg) : 0 ----------------
__global__ void a3t_dinv_kernel(const float* __restrict__ deg, float* __restrict__ dinv, int n) {
    int t = blockIdx.x * blockDim.x + threadIdx.x;
    if (t >= n) return;
    float d = deg[t];
    dinv[t] = (d > 0.0f) ? rsqrtf(d) : 0.0f;
}

// ---------------- agg[dst,:] += dinv[src]*w*dinv[dst] * x[src,:] ----------------
__global__ void a3t_scatter_kernel(const int* __restrict__ eidx, const float* __restrict__ w,
                                   const float* __restrict__ dinv, const float* __restrict__ x,
                                   float* __restrict__ agg, int E) {
    int e = blockIdx.x * blockDim.x + threadIdx.x;
    if (e >= E) return;
    int s = eidx[e];
    int d = eidx[E + e];
    float c = dinv[s] * w[e] * dinv[d];
    const float4* xr = (const float4*)(x + (size_t)s * P_DIM);   // 12 floats = 3x float4, 16B aligned
    float4 x0 = xr[0], x1 = xr[1], x2 = xr[2];
    float* ar = agg + (size_t)d * P_DIM;
    atomicAdd(ar + 0,  c * x0.x); atomicAdd(ar + 1,  c * x0.y);
    atomicAdd(ar + 2,  c * x0.z); atomicAdd(ar + 3,  c * x0.w);
    atomicAdd(ar + 4,  c * x1.x); atomicAdd(ar + 5,  c * x1.y);
    atomicAdd(ar + 6,  c * x1.z); atomicAdd(ar + 7,  c * x1.w);
    atomicAdd(ar + 8,  c * x2.x); atomicAdd(ar + 9,  c * x2.y);
    atomicAdd(ar + 10, c * x2.z); atomicAdd(ar + 11, c * x2.w);
}

// -------- collapse conv+linear: az[k]=cw_z.W_z[:,k]  bz[k]=cb_z.W_z[:,k]+b_z[k] (same for h-gate) --------
__global__ void a3t_coef_kernel(const float* __restrict__ cwz, const float* __restrict__ cbz,
                                const float* __restrict__ Wz,  const float* __restrict__ lbz,
                                const float* __restrict__ cwh, const float* __restrict__ cbh,
                                const float* __restrict__ Wh,  const float* __restrict__ lbh,
                                float* __restrict__ coefs) {
    int k = threadIdx.x;                       // 128 threads
    float az = 0.f, bz = 0.f, ah = 0.f, bh = 0.f;
    for (int h = 0; h < H_DIM; ++h) {
        float wz = Wz[h * H_DIM + k];          // lin_w is [2H,H] row-major, first H rows used
        float wh = Wh[h * H_DIM + k];
        az = fmaf(cwz[h], wz, az);
        bz = fmaf(cbz[h], wz, bz);
        ah = fmaf(cwh[h], wh, ah);
        bh = fmaf(cbh[h], wh, bh);
    }
    coefs[k]             = az;
    coefs[H_DIM + k]     = bz + lbz[k];
    coefs[2 * H_DIM + k] = ah;
    coefs[3 * H_DIM + k] = bh + lbh[k];
}

// ---------------- softmax over 12 attention values ----------------
__global__ void a3t_softmax_kernel(const float* __restrict__ att, float* __restrict__ probs) {
    if (threadIdx.x != 0) return;
    float m = att[0];
    for (int p = 1; p < P_DIM; ++p) m = fmaxf(m, att[p]);
    float e[P_DIM];
    float s = 0.0f;
    for (int p = 0; p < P_DIM; ++p) { e[p] = __expf(att[p] - m); s += e[p]; }
    float inv = 1.0f / s;
    for (int p = 0; p < P_DIM; ++p) probs[p] = e[p] * inv;
}

// ------- fused gates + attention sum + ELU: one block (128 threads) per node -------
__global__ void a3t_act_kernel(const float* __restrict__ agg, const float* __restrict__ x,
                               const float* __restrict__ dinv, const float* __restrict__ coefs,
                               const float* __restrict__ probs, _Float16* __restrict__ hbuf) {
    int n = blockIdx.x;
    int k = threadIdx.x;                       // 0..127
    __shared__ float sAgg[P_DIM];
    __shared__ float sProb[P_DIM];
    if (k < P_DIM) {
        float dv = dinv[n];
        // add self-loop term dinv[n]^2 * x[n,p]
        sAgg[k]  = agg[(size_t)n * P_DIM + k] + x[(size_t)n * P_DIM + k] * dv * dv;
        sProb[k] = probs[k];
    }
    __syncthreads();
    float az = coefs[k];
    float bz = coefs[H_DIM + k];
    float ah = coefs[2 * H_DIM + k];
    float bh = coefs[3 * H_DIM + k];
    float acc = 0.0f;
#pragma unroll
    for (int p = 0; p < P_DIM; ++p) {
        float s   = sAgg[p];
        float zp  = fmaf(s, az, bz);
        float hp  = fmaf(s, ah, bh);
        float sig = 1.0f / (1.0f + __expf(-zp));                           // sigmoid
        float th  = fmaf(2.0f, 1.0f / (1.0f + __expf(-2.0f * hp)), -1.0f); // tanh
        acc = fmaf(sProb[p] * (1.0f - sig), th, acc);
    }
    float hv = (acc > 0.0f) ? acc : (__expf(acc) - 1.0f);                  // ELU
    hbuf[(size_t)n * H_DIM + k] = (_Float16)hv;
}

// ------- pack lin_out_w [128,12] into WMMA B-operand lane layout (cols padded to 16) -------
// half index hi of lane holds B[K][N] with K = 32*kc + hi + 16*(lane>>4), N = lane&15
__global__ void a3t_bpack_kernel(const float* __restrict__ Wout, _Float16* __restrict__ Bpack) {
    int t = blockIdx.x * blockDim.x + threadIdx.x;
    if (t >= 4 * 32 * 16) return;
    int kc   = t >> 9;
    int rem  = t & 511;
    int lane = rem >> 4;
    int hi   = rem & 15;
    int K    = kc * 32 + hi + 16 * (lane >> 4);
    int col  = lane & 15;
    float v  = (col < O_DIM) ? Wout[K * O_DIM + col] : 0.0f;
    Bpack[(size_t)(kc * 32 + lane) * 16 + hi] = (_Float16)v;
}

// ------- out[N,12] = h[N,128] @ W[128,12] + b  via v_wmma_f32_16x16x32_f16 -------
// one wave32 per 16-node tile; K = 128 -> 4 chunks of 32.
// Epilogue stages the 16x12 tile (contiguous 768 B of `out`) through LDS so the
// writeback is 48 coalesced b128 stores with a single uniform bounds branch.
__global__ void a3t_gemm_kernel(const _Float16* __restrict__ hbuf,
                                const _Float16* __restrict__ Bpack,
                                const float* __restrict__ bias,
                                float* __restrict__ out, int N) {
    int lane = threadIdx.x;          // 0..31
    int tile = blockIdx.x;
    int half = lane >> 4;            // 0 or 1
    int mrow = lane & 15;

    // B operand: persistent in registers, identical for every wave
    v16h b0 = *(const v16h*)(Bpack + (size_t)(0 * 32 + lane) * 16);
    v16h b1 = *(const v16h*)(Bpack + (size_t)(1 * 32 + lane) * 16);
    v16h b2 = *(const v16h*)(Bpack + (size_t)(2 * 32 + lane) * 16);
    v16h b3 = *(const v16h*)(Bpack + (size_t)(3 * 32 + lane) * 16);

    int node = tile * 16 + mrow;
    const _Float16* arow = hbuf + (size_t)(node < N ? node : 0) * H_DIM;

    v8f c = {};
#pragma unroll
    for (int kc = 0; kc < 4; ++kc) {
        // A k order used here (contiguous, same as Bpack): lane<16 -> halves 0..15 = K 0..15
        //                          lane>=16 -> halves 0..15 = K 16..31
        int K0 = kc * 32;
        float4 lo = *(const float4*)(arow + K0 + 16 * half);       // 8 contiguous f16
        float4 hi = *(const float4*)(arow + K0 + 16 * half + 8);   // 8 contiguous f16
        v16h a;
        __builtin_memcpy(&a, &lo, 16);
        __builtin_memcpy((char*)&a + 16, &hi, 16);
        v16h bk = (kc == 0) ? b0 : (kc == 1) ? b1 : (kc == 2) ? b2 : b3;
        c = __builtin_amdgcn_wmma_f32_16x16x32_f16(false, a, false, bk, (short)0, c,
                                                   false, false);
    }

    // C/D layout: lane holds column N = lane&15, rows M = i + 8*(lane>>4).
    // Stage tile into LDS (bank-conflict-free: cols 0..11 x rows i,i+8 per store).
    __shared__ float sC[16 * O_DIM];
    if (mrow < O_DIM) {
        float bv = bias[mrow];
#pragma unroll
        for (int i = 0; i < 8; ++i)
            sC[(i + 8 * half) * O_DIM + mrow] = c[i] + bv;
    }
    __syncthreads();

    int base = tile * 16 * O_DIM;                 // dword offset of tile in out
    if (tile * 16 + 16 <= N) {
        // full tile: 192 contiguous dwords, 16B-aligned -> coalesced b128 stores
        float4* o4 = (float4*)(out + base);
        const float4* s4 = (const float4*)sC;
        o4[lane] = s4[lane];
        if (lane < 16) o4[32 + lane] = s4[32 + lane];
    } else {
        // tail tile (not hit for N % 16 == 0): guarded scalar stores
        for (int t = lane; t < 16 * O_DIM; t += 32) {
            int onode = tile * 16 + t / O_DIM;
            if (onode < N) out[base + t] = sC[t];
        }
    }
}

extern "C" void kernel_launch(void* const* d_in, const int* in_sizes, int n_in,
                              void* d_out, int out_size, void* d_ws, size_t ws_size,
                              hipStream_t stream) {
    const float* x    = (const float*)d_in[0];
    const int*   eidx = (const int*)  d_in[1];
    const float* ew   = (const float*)d_in[2];
    const float* att  = (const float*)d_in[3];
    const float* cwz  = (const float*)d_in[4];
    const float* cbz  = (const float*)d_in[5];
    const float* Wz   = (const float*)d_in[6];
    const float* lbz  = (const float*)d_in[7];
    // d_in[8..11] = r-gate params: provably dead (hidden state is zero)
    const float* cwh  = (const float*)d_in[12];
    const float* cbh  = (const float*)d_in[13];
    const float* Wh   = (const float*)d_in[14];
    const float* lbh  = (const float*)d_in[15];
    const float* Wout = (const float*)d_in[16];
    const float* bout = (const float*)d_in[17];
    float* out = (float*)d_out;

    const int N = in_sizes[0] / P_DIM;   // 20000
    const int E = in_sizes[2];           // 640000

    // workspace carve-out (256B aligned slices; ~6.25 MB total)
    char* ws = (char*)d_ws;
    size_t off = 0;
    auto carve = [&](size_t bytes) -> void* {
        void* p = ws + off;
        off = (off + bytes + 255) & ~(size_t)255;
        return p;
    };
    float*    deg   = (float*)   carve((size_t)N * sizeof(float));
    float*    dinv  = (float*)   carve((size_t)N * sizeof(float));
    float*    agg   = (float*)   carve((size_t)N * P_DIM * sizeof(float));
    float*    coefs = (float*)   carve(4 * H_DIM * sizeof(float));
    float*    probs = (float*)   carve(P_DIM * sizeof(float));
    _Float16* Bpack = (_Float16*)carve(4 * 32 * 16 * sizeof(_Float16));
    _Float16* hbuf  = (_Float16*)carve((size_t)N * H_DIM * sizeof(_Float16));
    (void)ws_size; (void)n_in; (void)out_size;

    const int T = 256;
    a3t_init_kernel   <<<(N * P_DIM + T - 1) / T, T, 0, stream>>>(agg, deg, N);
    a3t_degree_kernel <<<(E + T - 1) / T,          T, 0, stream>>>(eidx, ew, deg, E);
    a3t_dinv_kernel   <<<(N + T - 1) / T,          T, 0, stream>>>(deg, dinv, N);
    a3t_scatter_kernel<<<(E + T - 1) / T,          T, 0, stream>>>(eidx, ew, dinv, x, agg, E);
    a3t_coef_kernel   <<<1, H_DIM, 0, stream>>>(cwz, cbz, Wz, lbz, cwh, cbh, Wh, lbh, coefs);
    a3t_softmax_kernel<<<1, 32, 0, stream>>>(att, probs);
    a3t_act_kernel    <<<N, H_DIM, 0, stream>>>(agg, x, dinv, coefs, probs, hbuf);
    a3t_bpack_kernel  <<<(4 * 32 * 16 + T - 1) / T, T, 0, stream>>>(Wout, Bpack);
    int nTiles = (N + 15) / 16;          // 1250
    a3t_gemm_kernel   <<<nTiles, 32, 0, stream>>>(hbuf, Bpack, bout, out, N);
}
